// VIEW2_9345848836755
// MI455X (gfx1250) — compile-verified, matrix operand order checked by reading, not run on hardware
//
#include <hip/hip_runtime.h>

#define N_NODES 20000
#define N_EDGES 640000
#define D 256
#define MTILES (N_NODES / 16)   // 1250
#define DEG_PAD 20480

typedef __attribute__((ext_vector_type(16))) _Float16 v16h;
typedef __attribute__((ext_vector_type(8)))  float    v8f;

union Frag {
    v16h v;
    uint4 u[2];
};

// ---------------- utility kernels ----------------

__global__ void k_zero_f32(float* __restrict__ p, int n) {
    int i = blockIdx.x * blockDim.x + threadIdx.x;
    if (i < n) p[i] = 0.0f;
}

__global__ void k_degree(const int* __restrict__ src, const int* __restrict__ dst,
                         float* __restrict__ deg_out, float* __restrict__ deg_in, int ne) {
    int i = blockIdx.x * blockDim.x + threadIdx.x;
    if (i < ne) {
        atomicAdd(deg_out + src[i], 1.0f);
        atomicAdd(deg_in  + dst[i], 1.0f);
    }
}

__global__ void k_rsqrt_inplace(float* __restrict__ p, int n) {
    int i = blockIdx.x * blockDim.x + threadIdx.x;
    if (i < n) {
        float d = p[i];
        d = d < 1.0f ? 1.0f : d;
        p[i] = rsqrtf(d);
    }
}

// Wt[n*D + k] = (f16) W[k*D + n]  (col-major f16 copy of W, so the B-fragment
// loads use the same contiguous-16B pattern as the A fragment)
__global__ void k_transpose_w(const float* __restrict__ W, _Float16* __restrict__ Wt) {
    int i = blockIdx.x * blockDim.x + threadIdx.x;   // D*D
    int n = i >> 8, k = i & 255;
    Wt[n * D + k] = (_Float16)W[k * D + n];
}

// one wave per edge; lane handles 8 f32 at 4*lane and 128+4*lane (two float4 halves of the row)
__global__ void k_scatter_f32(const float* __restrict__ x,
                              const int* __restrict__ src, const int* __restrict__ dst,
                              const float* __restrict__ rs_out, float* __restrict__ acc) {
    int gid = blockIdx.x * blockDim.x + threadIdx.x;
    int e = gid >> 5, lane = gid & 31;
    if (e >= N_EDGES) return;
    int s = src[e], d = dst[e];
    float sc = rs_out[s];
    const float4* xp = (const float4*)(x + (size_t)s * D);
    float* ap = acc + (size_t)d * D;
    float4 v0 = xp[lane];        // bytes [0,512)  coalesced
    float4 v1 = xp[lane + 32];   // bytes [512,1024) coalesced
    int o = lane * 4;
    atomicAdd(ap + o + 0, v0.x * sc);
    atomicAdd(ap + o + 1, v0.y * sc);
    atomicAdd(ap + o + 2, v0.z * sc);
    atomicAdd(ap + o + 3, v0.w * sc);
    atomicAdd(ap + 128 + o + 0, v1.x * sc);
    atomicAdd(ap + 128 + o + 1, v1.y * sc);
    atomicAdd(ap + 128 + o + 2, v1.z * sc);
    atomicAdd(ap + 128 + o + 3, v1.w * sc);
}

// same, but the source matrix is f16 (layer-2 hidden state)
__global__ void k_scatter_f16(const _Float16* __restrict__ h,
                              const int* __restrict__ src, const int* __restrict__ dst,
                              const float* __restrict__ rs_out, float* __restrict__ acc) {
    int gid = blockIdx.x * blockDim.x + threadIdx.x;
    int e = gid >> 5, lane = gid & 31;
    if (e >= N_EDGES) return;
    int s = src[e], d = dst[e];
    float sc = rs_out[s];
    uint4 raw = *(const uint4*)(h + (size_t)s * D + lane * 8);  // 8 halves, coalesced
    const _Float16* hv = (const _Float16*)&raw;
    float* ap = acc + (size_t)d * D + lane * 8;
#pragma unroll
    for (int i = 0; i < 8; ++i)
        atomicAdd(ap + i, (float)hv[i] * sc);
}

__global__ void k_norm_to_f16(const float* __restrict__ acc, const float* __restrict__ rs_in,
                              _Float16* __restrict__ out) {
    int i = blockIdx.x * blockDim.x + threadIdx.x;
    if (i < N_NODES * D)
        out[i] = (_Float16)(acc[i] * rs_in[i >> 8]);
}

// ---------------- dual-relation WMMA GEMM ----------------
// out[M,256] = opt_relu( Aa@Wa^T_stored + Ab@Wb^T_stored + ba + bb )
// One wave computes a 16x64 slab (4 n-tiles); both relations accumulate into
// the same f32 C fragments. K loop: 8 chunks of 32 -> 64 v_wmma per wave.
template <bool RELU, typename OutT>
__global__ __launch_bounds__(256) void
k_gemm_dual(const _Float16* __restrict__ Aa, const _Float16* __restrict__ Ab,
            const _Float16* __restrict__ Wa, const _Float16* __restrict__ Wb,
            const float* __restrict__ ba, const float* __restrict__ bb,
            OutT* __restrict__ out) {
    const int lane  = threadIdx.x & 31;
    const int wave  = (blockIdx.x * blockDim.x + threadIdx.x) >> 5;
    const int mtile = wave >> 2;        // 0..1249
    const int ngrp  = wave & 3;         // 64-column group
    const int lm    = lane & 15;
    const int kg8   = ((lane >> 4) & 1) * 8;   // K sub-group offset per 16-bit A layout
    const int rowA  = mtile * 16 + lm;
    const int nbase = ngrp * 64;

    v8f c[4];
#pragma unroll
    for (int t = 0; t < 4; ++t) {
        float bs = ba[nbase + t * 16 + lm] + bb[nbase + t * 16 + lm];
#pragma unroll
        for (int r = 0; r < 8; ++r) c[t][r] = bs;
    }

    for (int kc = 0; kc < D; kc += 32) {
        Frag a0, a1;
        const _Float16* pa0 = Aa + (size_t)rowA * D + kc + kg8;
        a0.u[0] = *(const uint4*)pa0;
        a0.u[1] = *(const uint4*)(pa0 + 16);
        const _Float16* pa1 = Ab + (size_t)rowA * D + kc + kg8;
        a1.u[0] = *(const uint4*)pa1;
        a1.u[1] = *(const uint4*)(pa1 + 16);
#pragma unroll
        for (int t = 0; t < 4; ++t) {
            const int nrow = nbase + t * 16 + lm;   // column of output = row of Wt
            Frag w0, w1;
            const _Float16* pw0 = Wa + (size_t)nrow * D + kc + kg8;
            w0.u[0] = *(const uint4*)pw0;
            w0.u[1] = *(const uint4*)(pw0 + 16);
            const _Float16* pw1 = Wb + (size_t)nrow * D + kc + kg8;
            w1.u[0] = *(const uint4*)pw1;
            w1.u[1] = *(const uint4*)(pw1 + 16);
            c[t] = __builtin_amdgcn_wmma_f32_16x16x32_f16(
                false, a0.v, false, w0.v, (short)0, c[t], false, false);
            c[t] = __builtin_amdgcn_wmma_f32_16x16x32_f16(
                false, a1.v, false, w1.v, (short)0, c[t], false, false);
        }
    }

    const int mbase = mtile * 16 + ((lane >> 4) << 3);
#pragma unroll
    for (int t = 0; t < 4; ++t) {
        const int nc = nbase + t * 16 + lm;
#pragma unroll
        for (int r = 0; r < 8; ++r) {
            float v = c[t][r];
            if (RELU) v = v > 0.0f ? v : 0.0f;
            out[(size_t)(mbase + r) * D + nc] = (OutT)v;
        }
    }
}

// ---------------- launch ----------------

extern "C" void kernel_launch(void* const* d_in, const int* in_sizes, int n_in,
                              void* d_out, int out_size, void* d_ws, size_t ws_size,
                              hipStream_t stream) {
    (void)in_sizes; (void)n_in; (void)out_size; (void)ws_size;

    const float* x        = (const float*)d_in[0];
    const int*   side_src = (const int*)d_in[1];
    const int*   side_dst = (const int*)d_in[2];
    const int*   upd_src  = (const int*)d_in[3];
    const int*   upd_dst  = (const int*)d_in[4];
    const float* W1s = (const float*)d_in[5];
    const float* b1s = (const float*)d_in[6];
    const float* W1u = (const float*)d_in[7];
    const float* b1u = (const float*)d_in[8];
    const float* W2s = (const float*)d_in[9];
    const float* b2s = (const float*)d_in[10];
    const float* W2u = (const float*)d_in[11];
    const float* b2u = (const float*)d_in[12];
    float* out = (float*)d_out;

    char* wsp = (char*)d_ws;
    auto take = [&](size_t bytes) {
        char* p = wsp;
        wsp += (bytes + 255) & ~(size_t)255;
        return p;
    };
    float* degs  = (float*)take((size_t)4 * DEG_PAD * sizeof(float));
    float* rs_os = degs;                 // becomes rsqrt(max(deg,1)) in place
    float* rs_is = degs + DEG_PAD;
    float* rs_ou = degs + 2 * DEG_PAD;
    float* rs_iu = degs + 3 * DEG_PAD;
    float* acc   = (float*)take(2 * (size_t)N_NODES * D * sizeof(float));
    float* acc_s = acc;
    float* acc_u = acc + (size_t)N_NODES * D;
    _Float16* agg_s = (_Float16*)take((size_t)N_NODES * D * 2);
    _Float16* agg_u = (_Float16*)take((size_t)N_NODES * D * 2);
    _Float16* hbuf  = (_Float16*)take((size_t)N_NODES * D * 2);
    _Float16* Wt1s  = (_Float16*)take((size_t)D * D * 2);
    _Float16* Wt1u  = (_Float16*)take((size_t)D * D * 2);
    _Float16* Wt2s  = (_Float16*)take((size_t)D * D * 2);
    _Float16* Wt2u  = (_Float16*)take((size_t)D * D * 2);

    const int TB = 256;
    const int ND = N_NODES * D;                    // 5,120,000
    const int edge_blocks = (N_EDGES * 32) / TB;   // 80,000
    const int gemm_blocks = (MTILES * 4) / 8;      // 625

    // degrees (shared by both layers)
    k_zero_f32<<<(4 * DEG_PAD) / TB, TB, 0, stream>>>(degs, 4 * DEG_PAD);
    k_degree<<<(N_EDGES + TB - 1) / TB, TB, 0, stream>>>(side_src, side_dst, rs_os, rs_is, N_EDGES);
    k_degree<<<(N_EDGES + TB - 1) / TB, TB, 0, stream>>>(upd_src, upd_dst, rs_ou, rs_iu, N_EDGES);
    k_rsqrt_inplace<<<(4 * DEG_PAD) / TB, TB, 0, stream>>>(degs, 4 * DEG_PAD);

    // weights -> col-major f16
    k_transpose_w<<<(D * D) / TB, TB, 0, stream>>>(W1s, Wt1s);
    k_transpose_w<<<(D * D) / TB, TB, 0, stream>>>(W1u, Wt1u);
    k_transpose_w<<<(D * D) / TB, TB, 0, stream>>>(W2s, Wt2s);
    k_transpose_w<<<(D * D) / TB, TB, 0, stream>>>(W2u, Wt2u);

    // ---- layer 1 ----
    k_zero_f32<<<(2 * ND) / TB, TB, 0, stream>>>(acc, 2 * ND);
    k_scatter_f32<<<edge_blocks, TB, 0, stream>>>(x, side_src, side_dst, rs_os, acc_s);
    k_scatter_f32<<<edge_blocks, TB, 0, stream>>>(x, upd_src, upd_dst, rs_ou, acc_u);
    k_norm_to_f16<<<ND / TB, TB, 0, stream>>>(acc_s, rs_is, agg_s);
    k_norm_to_f16<<<ND / TB, TB, 0, stream>>>(acc_u, rs_iu, agg_u);
    k_gemm_dual<true, _Float16><<<gemm_blocks, TB, 0, stream>>>(
        agg_s, agg_u, Wt1s, Wt1u, b1s, b1u, hbuf);

    // ---- layer 2 ----
    k_zero_f32<<<(2 * ND) / TB, TB, 0, stream>>>(acc, 2 * ND);
    k_scatter_f16<<<edge_blocks, TB, 0, stream>>>(hbuf, side_src, side_dst, rs_os, acc_s);
    k_scatter_f16<<<edge_blocks, TB, 0, stream>>>(hbuf, upd_src, upd_dst, rs_ou, acc_u);
    k_norm_to_f16<<<ND / TB, TB, 0, stream>>>(acc_s, rs_is, agg_s);
    k_norm_to_f16<<<ND / TB, TB, 0, stream>>>(acc_u, rs_iu, agg_u);
    k_gemm_dual<false, float><<<gemm_blocks, TB, 0, stream>>>(
        agg_s, agg_u, Wt2s, Wt2u, b2s, b2u, out);
}
